// MOTGraphModel_74302934221142
// MI455X (gfx1250) — matrix operand order repaired and checked by reading, not compiled, not run on hardware
//
#include <hip/hip_runtime.h>
#include <hip/hip_bf16.h>

#define NN 10000
#define EE 320000

typedef __attribute__((ext_vector_type(16))) _Float16 v16h;
typedef __attribute__((ext_vector_type(8)))  _Float16 v8h;
typedef __attribute__((ext_vector_type(8)))  float    v8f;

// ---------------- WMMA helpers (CDNA5 16x16x32 f16, f32 accum) ----------------

__device__ __forceinline__ v8f wmma32(const v16h& a, const v16h& b, const v8f& c) {
  // (neg_a, A, neg_b, B, c_mod, C, reuse_a, reuse_b)
  return __builtin_amdgcn_wmma_f32_16x16x32_f16(false, a, false, b, (short)0, c, false, false);
}

// Generic (scalar-gather) A fragment: lane holds row m = lane&15.
// lanes 0-15: elems 0..7 -> K 0..7,  elems 8..15 -> K 16..23
// lanes 16-31: elems 0..7 -> K 8..15, elems 8..15 -> K 24..31
template <typename F>
__device__ __forceinline__ v16h loadA(F f) {
  const int lane = threadIdx.x & 31;
  const int ml = lane & 15;
  const int kh = (lane >> 4) << 3;  // 0 or 8
  v16h a;
#pragma unroll
  for (int j = 0; j < 8; ++j) {
    a[j]     = f(ml, kh + j);
    a[j + 8] = f(ml, kh + j + 16);
  }
  return a;
}

// Vectorized A fragment from a 32-f16 contiguous, 16B-aligned row chunk
// (row pointer must already be this lane's row, i.e. computed with lane&15).
__device__ __forceinline__ v16h loadA_row32(const _Float16* row) {
  const int kh = ((threadIdx.x & 31) >> 4) << 3;  // 0 or 8
  v8h lo = *(const v8h*)(row + kh);
  v8h hi = *(const v8h*)(row + kh + 16);
  v16h a;
#pragma unroll
  for (int j = 0; j < 8; ++j) { a[j] = lo[j]; a[j + 8] = hi[j]; }
  return a;
}

// Same but source is f32 (converted on the fly).
__device__ __forceinline__ v16h loadA_row32_f32(const float* row) {
  const int kh = ((threadIdx.x & 31) >> 4) << 3;
  v8f lo = *(const v8f*)(row + kh);
  v8f hi = *(const v8f*)(row + kh + 16);
  v16h a;
#pragma unroll
  for (int j = 0; j < 8; ++j) { a[j] = (_Float16)lo[j]; a[j + 8] = (_Float16)hi[j]; }
  return a;
}

// B fragment from fragment-native weight storage: 16 contiguous f16 per lane.
// t = nt*Ktiles + ki.
__device__ __forceinline__ v16h loadBfrag(const _Float16* Wf, int t) {
  return *(const v16h*)(Wf + ((size_t)t << 9) + ((threadIdx.x & 31) << 4));
}

// C/D 16x16 f32: lane holds col n = lane&15; VGPR r holds row r (+8 for lanes>=16).
template <typename F>
__device__ __forceinline__ void storeC(const v8f& c, F f) {
  const int rh = ((threadIdx.x & 31) >> 4) << 3;
#pragma unroll
  for (int r = 0; r < 8; ++r) f(rh + r, c[r]);
}

// ---------------- Weight prep: f32 -> f16 fragment-native layouts ----------------
// For W[K][Nf]: frag element idx -> e=idx&15, lane=(idx>>4)&31, t=idx>>9,
// ki=t%Ktiles, nt=t/Ktiles, K = ki*32 + ((e>=8)?16:0) + ((lane>=16)?8:0) + (e&7), N = nt*16 + (lane&15).
#define PREP_TOTAL 227328

__device__ __forceinline__ void frag_kn(int idx, int Ktiles, int& k, int& n) {
  const int e = idx & 15;
  const int lane = (idx >> 4) & 31;
  const int t = idx >> 9;
  const int ki = t % Ktiles;
  const int nt = t / Ktiles;
  k = ki * 32 + ((e >> 3) << 4) + ((lane >> 4) << 3) + (e & 7);
  n = nt * 16 + (lane & 15);
}

__global__ void prep_weights(
    const float* __restrict__ cw0, const float* __restrict__ cw,
    const float* __restrict__ ie_w1, const float* __restrict__ ie_w2,
    const float* __restrict__ in_w1, const float* __restrict__ in_w2,
    const float* __restrict__ em_w1, const float* __restrict__ em_w2,
    _Float16* __restrict__ Wc0F, _Float16* __restrict__ WcF,
    _Float16* __restrict__ Wie1F, _Float16* __restrict__ Wie2F,
    _Float16* __restrict__ Win1F, _Float16* __restrict__ Win2F,
    _Float16* __restrict__ Wem1F, _Float16* __restrict__ Wem2F)
{
  int i = blockIdx.x * blockDim.x + threadIdx.x;
  const int stride = gridDim.x * blockDim.x;
  for (; i < PREP_TOTAL; i += stride) {
    int idx = i, k, n;
    if (idx < 2048) {  // Wc0: Ksrc=12 (pad 32), Nf=64, k = c*4+dy*2+dx
      frag_kn(idx, 1, k, n);
      Wc0F[idx] = (k < 12) ? (_Float16)cw0[n * 12 + k] : (_Float16)0.f;
      continue;
    }
    idx -= 2048;
    if (idx < 81920) {  // Wc: 5 layers, K=256 (k = dydx*64 + ci), Nf=64
      const int l = idx / 16384, sub = idx % 16384;
      frag_kn(sub, 8, k, n);
      const int dydx = k >> 6, ci = k & 63;
      WcF[idx] = (_Float16)cw[(((l * 64 + n) * 64 + ci) * 4) + dydx];
      continue;
    }
    idx -= 81920;
    if (idx < 49152) {  // Wie1: 4 layers, K=192
      const int l = idx / 12288, sub = idx % 12288;
      frag_kn(sub, 6, k, n);
      Wie1F[idx] = (_Float16)ie_w1[((size_t)l * 192 + k) * 64 + n];
      continue;
    }
    idx -= 49152;
    if (idx < 16384) {  // Wie2: 4 layers, K=64
      const int l = idx / 4096, sub = idx % 4096;
      frag_kn(sub, 2, k, n);
      Wie2F[idx] = (_Float16)ie_w2[((size_t)l * 64 + k) * 64 + n];
      continue;
    }
    idx -= 16384;
    if (idx < 32768) {  // Win1: 4 layers, K=128
      const int l = idx / 8192, sub = idx % 8192;
      frag_kn(sub, 4, k, n);
      Win1F[idx] = (_Float16)in_w1[((size_t)l * 128 + k) * 64 + n];
      continue;
    }
    idx -= 32768;
    if (idx < 16384) {  // Win2: 4 layers, K=64
      const int l = idx / 4096, sub = idx % 4096;
      frag_kn(sub, 2, k, n);
      Win2F[idx] = (_Float16)in_w2[((size_t)l * 64 + k) * 64 + n];
      continue;
    }
    idx -= 16384;
    if (idx < 20480) {  // Wem1: Ksrc=134 (pad 160), Nf=128
      frag_kn(idx, 5, k, n);
      Wem1F[idx] = (k < 134) ? (_Float16)em_w1[k * 128 + n] : (_Float16)0.f;
      continue;
    }
    idx -= 20480;
    frag_kn(idx, 4, k, n);  // Wem2: K=128, Nf=64
    Wem2F[idx] = (_Float16)em_w2[k * 64 + n];
  }
}

// ---------------- Fused 6-layer conv stack + node_update per image ----------------
__global__ __launch_bounds__(256) void conv_node_kernel(
    const float* __restrict__ crop, const float* __restrict__ xin,
    const float* __restrict__ cb0, const float* __restrict__ cb,
    const float* __restrict__ nu_w, const float* __restrict__ nu_b,
    const float* __restrict__ nu_g, const float* __restrict__ nu_be,
    const _Float16* __restrict__ Wc0F, const _Float16* __restrict__ WcF,
    _Float16* __restrict__ nodeF16)
{
  __shared__ __align__(32) _Float16 sA[16384];  // 32KB multi-use
  __shared__ __align__(32) _Float16 sB[16384];  // 32KB layer1 [16][16][64]
  const int img = blockIdx.x;
  const int tid = threadIdx.x;
  const int w = tid >> 5, lane = tid & 31;
  const float* cimg = crop + (size_t)img * 3 * 64 * 64;
  float* sFeat = (float*)(sA + 6144);   // 64 floats (crop_feat)
  float* sH    = (float*)(sA + 6400);   // 64 floats (node hidden)
  float* sStat = (float*)(sA + 6528);   // mu, inv-sigma

  for (int q = 0; q < 4; ++q) {
    const int qy = q >> 1, qx = q & 1;
    __syncthreads();
    // conv0 quadrant: M=256 (16x16 pos), K=12->32, N=64 -> sA[pos][64]
    for (int t = w; t < 64; t += 8) {
      const int mt = t >> 2, nt = t & 3;
      v16h a = loadA([&](int ml, int k) -> _Float16 {
        if (k >= 12) return (_Float16)0.f;
        const int m = mt * 16 + ml;
        const int y = qy * 16 + (m >> 4), xx = qx * 16 + (m & 15);
        const int c = k >> 2, dy = (k >> 1) & 1, dx = k & 1;
        return (_Float16)cimg[(c * 64 + (2 * y + dy)) * 64 + (2 * xx + dx)];
      });
      v16h b = loadBfrag(Wc0F, nt);
      v8f c = {};
      c = wmma32(a, b, c);
      const int n = nt * 16 + (lane & 15);
      const float bias = cb0[n];
      storeC(c, [&](int r, float v) {
        const int m = mt * 16 + r;
        float h = v + bias;
        sA[m * 64 + n] = (_Float16)(h > 0.f ? h : 0.f);
      });
    }
    __syncthreads();
    // conv1: M=64 (8x8), K=256 (im2col of sA, 32-f16 contiguous runs), N=64 -> sB
    {
      const int nt = w & 3;
      v16h kb[8];
#pragma unroll
      for (int ki = 0; ki < 8; ++ki) kb[ki] = loadBfrag(WcF, nt * 8 + ki);
      for (int ms = 0; ms < 2; ++ms) {
        const int mt = (w >> 2) + 2 * ms;
        const int m = mt * 16 + (lane & 15);
        const int y = m >> 3, xx = m & 7;
        v8f c = {};
#pragma unroll
        for (int ki = 0; ki < 8; ++ki) {
          const int dydx = ki >> 1;
          const _Float16* row =
              sA + ((2 * y + (dydx >> 1)) * 16 + (2 * xx + (dydx & 1))) * 64 + (ki & 1) * 32;
          c = wmma32(loadA_row32(row), kb[ki], c);
        }
        const int n = nt * 16 + (lane & 15);
        const float bias = cb[n];
        storeC(c, [&](int r, float v) {
          const int mm = mt * 16 + r;
          const int yy = mm >> 3, xo = mm & 7;
          float h = v + bias;
          sB[((qy * 8 + yy) * 16 + (qx * 8 + xo)) * 64 + n] = (_Float16)(h > 0.f ? h : 0.f);
        });
      }
    }
  }
  __syncthreads();
  // conv2: M=64 (8x8), K=256 from sB -> out2 = sA[0..4095]
  {
    const int nt = w & 3;
    const _Float16* W2 = WcF + 1 * 16384;
    v16h kb[8];
#pragma unroll
    for (int ki = 0; ki < 8; ++ki) kb[ki] = loadBfrag(W2, nt * 8 + ki);
    for (int ms = 0; ms < 2; ++ms) {
      const int mt = (w >> 2) + 2 * ms;
      const int m = mt * 16 + (lane & 15);
      const int y = m >> 3, xx = m & 7;
      v8f c = {};
#pragma unroll
      for (int ki = 0; ki < 8; ++ki) {
        const int dydx = ki >> 1;
        const _Float16* row =
            sB + ((2 * y + (dydx >> 1)) * 16 + (2 * xx + (dydx & 1))) * 64 + (ki & 1) * 32;
        c = wmma32(loadA_row32(row), kb[ki], c);
      }
      const int n = nt * 16 + (lane & 15);
      const float bias = cb[64 + n];
      storeC(c, [&](int r, float v) {
        const int mm = mt * 16 + r;
        float h = v + bias;
        sA[mm * 64 + n] = (_Float16)(h > 0.f ? h : 0.f);  // pos = y*8+x
      });
    }
  }
  __syncthreads();
  // conv3: M=16 (4x4), input out2 -> out3 = sA[4096..]
  if (w < 4) {
    const int nt = w;
    const _Float16* W3 = WcF + 2 * 16384;
    const int ml = lane & 15;
    const int y = ml >> 2, xx = ml & 3;
    v8f c = {};
#pragma unroll
    for (int ki = 0; ki < 8; ++ki) {
      const int dydx = ki >> 1;
      const _Float16* row =
          sA + ((2 * y + (dydx >> 1)) * 8 + (2 * xx + (dydx & 1))) * 64 + (ki & 1) * 32;
      c = wmma32(loadA_row32(row), loadBfrag(W3, nt * 8 + ki), c);
    }
    const int n = nt * 16 + ml;
    const float bias = cb[2 * 64 + n];
    storeC(c, [&](int r, float v) {
      float h = v + bias;
      sA[4096 + r * 64 + n] = (_Float16)(h > 0.f ? h : 0.f);
    });
  }
  __syncthreads();
  // conv4: M=4 (2x2) padded to 16 (rows clamped), input out3 -> out4 = sA[5120..]
  if (w < 4) {
    const int nt = w;
    const _Float16* W4 = WcF + 3 * 16384;
    const int mc = lane & 3;  // clamp padded rows onto valid data
    const int y = mc >> 1, xx = mc & 1;
    v8f c = {};
#pragma unroll
    for (int ki = 0; ki < 8; ++ki) {
      const int dydx = ki >> 1;
      const _Float16* row =
          sA + 4096 + ((2 * y + (dydx >> 1)) * 4 + (2 * xx + (dydx & 1))) * 64 + (ki & 1) * 32;
      c = wmma32(loadA_row32(row), loadBfrag(W4, nt * 8 + ki), c);
    }
    const int n = nt * 16 + (lane & 15);
    const float bias = cb[3 * 64 + n];
    storeC(c, [&](int r, float v) {
      if (r < 4) {
        float h = v + bias;
        sA[5120 + r * 64 + n] = (_Float16)(h > 0.f ? h : 0.f);
      }
    });
  }
  __syncthreads();
  // conv5: M=1 (all padded rows read the same valid data) -> crop_feat f32
  if (w < 4) {
    const int nt = w;
    const _Float16* W5 = WcF + 4 * 16384;
    v8f c = {};
#pragma unroll
    for (int ki = 0; ki < 8; ++ki) {
      const _Float16* row = sA + 5120 + (ki >> 1) * 64 + (ki & 1) * 32;  // pos = dydx
      c = wmma32(loadA_row32(row), loadBfrag(W5, nt * 8 + ki), c);
    }
    if (lane < 16) {  // row 0 lives in VGPR0 of lanes 0..15
      const int n = nt * 16 + lane;
      float h = c[0] + cb[4 * 64 + n];
      sFeat[n] = h > 0.f ? h : 0.f;
    }
  }
  __syncthreads();
  // node_update: relu(concat(x, crop_feat) @ nu_w + nu_b) then LayerNorm
  if (tid < 64) {
    float acc = nu_b[tid];
    const float* xr = xin + (size_t)img * 32;
#pragma unroll
    for (int k = 0; k < 32; ++k) acc += xr[k] * nu_w[k * 64 + tid];
#pragma unroll
    for (int k = 0; k < 64; ++k) acc += sFeat[k] * nu_w[(32 + k) * 64 + tid];
    sH[tid] = acc > 0.f ? acc : 0.f;
  }
  __syncthreads();
  if (tid == 0) {
    float mu = 0.f;
    for (int k = 0; k < 64; ++k) mu += sH[k];
    mu *= (1.f / 64.f);
    float var = 0.f;
    for (int k = 0; k < 64; ++k) { float d = sH[k] - mu; var += d * d; }
    var *= (1.f / 64.f);
    sStat[0] = mu;
    sStat[1] = rsqrtf(var + 1e-5f);
  }
  __syncthreads();
  if (tid < 64) {
    float v = (sH[tid] - sStat[0]) * sStat[1] * nu_g[tid] + nu_be[tid];
    nodeF16[(size_t)img * 64 + tid] = (_Float16)v;
  }
}

// ---------------- edge_update: LN(relu(edge_attr @ eu_w + b)) -> f16 ----------------
__global__ __launch_bounds__(256) void edge_update_kernel(
    const float* __restrict__ edge_attr, const float* __restrict__ eu_w,
    const float* __restrict__ eu_b, const float* __restrict__ eu_g,
    const float* __restrict__ eu_be, _Float16* __restrict__ edgeF16)
{
  const int e = blockIdx.x * blockDim.x + threadIdx.x;
  if (e >= EE) return;
  float a[6];
#pragma unroll
  for (int k = 0; k < 6; ++k) a[k] = edge_attr[(size_t)e * 6 + k];
  float h[64];
  float mu = 0.f;
#pragma unroll
  for (int o = 0; o < 64; ++o) {
    float acc = eu_b[o];
#pragma unroll
    for (int k = 0; k < 6; ++k) acc += a[k] * eu_w[k * 64 + o];
    acc = acc > 0.f ? acc : 0.f;
    h[o] = acc;
    mu += acc;
  }
  mu *= (1.f / 64.f);
  float var = 0.f;
#pragma unroll
  for (int o = 0; o < 64; ++o) { float d = h[o] - mu; var += d * d; }
  const float is = rsqrtf(var * (1.f / 64.f) + 1e-5f);
#pragma unroll
  for (int o = 0; o < 64; ++o)
    edgeF16[(size_t)e * 64 + o] = (_Float16)((h[o] - mu) * is * eu_g[o] + eu_be[o]);
}

// ---------------- interaction: message MLP + scatter-add (one wave / 16 edges) ----------------
__global__ __launch_bounds__(32) void msg_kernel(
    const _Float16* __restrict__ nodeF16, const _Float16* __restrict__ edgeF16,
    const int* __restrict__ ei,
    const _Float16* __restrict__ W1F, const float* __restrict__ b1,
    const _Float16* __restrict__ W2F, const float* __restrict__ b2,
    float* __restrict__ agg)
{
  __shared__ __align__(32) _Float16 sH[16 * 64];
  const int lane = threadIdx.x;
  const int e0 = blockIdx.x * 16;
  const int eRow = e0 + (lane & 15);
  __builtin_prefetch(W1F, 0, 3);
  __builtin_prefetch(W2F, 0, 3);
  const int s = ei[eRow];
  const int d = ei[EE + eRow];
  const _Float16* nd = nodeF16 + (size_t)d * 64;
  const _Float16* ns = nodeF16 + (size_t)s * 64;
  const _Float16* ee = edgeF16 + (size_t)eRow * 64;

  v16h af[6];  // A = [node[dst] | node[src] | edge], K=192
  af[0] = loadA_row32(nd);  af[1] = loadA_row32(nd + 32);
  af[2] = loadA_row32(ns);  af[3] = loadA_row32(ns + 32);
  af[4] = loadA_row32(ee);  af[5] = loadA_row32(ee + 32);
#pragma unroll
  for (int nt = 0; nt < 4; ++nt) {
    v8f c = {};
#pragma unroll
    for (int ki = 0; ki < 6; ++ki) c = wmma32(af[ki], loadBfrag(W1F, nt * 6 + ki), c);
    const int n = nt * 16 + (lane & 15);
    const float bias = b1[n];
    storeC(c, [&](int r, float v) {
      float h = v + bias;
      sH[r * 64 + n] = (_Float16)(h > 0.f ? h : 0.f);
    });
  }
  __syncthreads();
  v16h a2[2];
  a2[0] = loadA_row32(sH + (lane & 15) * 64);
  a2[1] = loadA_row32(sH + (lane & 15) * 64 + 32);
  int dr[8];
  const int rh = (lane >> 4) << 3;
#pragma unroll
  for (int r = 0; r < 8; ++r) dr[r] = ei[EE + e0 + rh + r];
#pragma unroll
  for (int nt = 0; nt < 4; ++nt) {
    v8f c = {};
#pragma unroll
    for (int ki = 0; ki < 2; ++ki) c = wmma32(a2[ki], loadBfrag(W2F, nt * 2 + ki), c);
    const int n = nt * 16 + (lane & 15);
    const float bias = b2[n];
#pragma unroll
    for (int r = 0; r < 8; ++r)
      unsafeAtomicAdd(&agg[(size_t)dr[r] * 64 + n], c[r] + bias);
  }
}

// ---------------- interaction: node MLP (in-place node update) ----------------
__global__ __launch_bounds__(32) void node_mlp_kernel(
    _Float16* __restrict__ nodeF16, const float* __restrict__ agg,
    const _Float16* __restrict__ W1F, const float* __restrict__ b1,
    const _Float16* __restrict__ W2F, const float* __restrict__ b2)
{
  __shared__ __align__(32) _Float16 sH[16 * 64];
  const int lane = threadIdx.x;
  const int r0 = blockIdx.x * 16;
  const int row = r0 + (lane & 15);
  __builtin_prefetch(W1F, 0, 3);
  __builtin_prefetch(W2F, 0, 3);
  const _Float16* np = nodeF16 + (size_t)row * 64;
  const float* ap = agg + (size_t)row * 64;
  v16h af[4];  // A = [node | agg], K=128
  af[0] = loadA_row32(np);        af[1] = loadA_row32(np + 32);
  af[2] = loadA_row32_f32(ap);    af[3] = loadA_row32_f32(ap + 32);
#pragma unroll
  for (int nt = 0; nt < 4; ++nt) {
    v8f c = {};
#pragma unroll
    for (int ki = 0; ki < 4; ++ki) c = wmma32(af[ki], loadBfrag(W1F, nt * 4 + ki), c);
    const int n = nt * 16 + (lane & 15);
    const float bias = b1[n];
    storeC(c, [&](int r, float v) {
      float h = v + bias;
      sH[r * 64 + n] = (_Float16)(h > 0.f ? h : 0.f);
    });
  }
  __syncthreads();
  v16h a2[2];
  a2[0] = loadA_row32(sH + (lane & 15) * 64);
  a2[1] = loadA_row32(sH + (lane & 15) * 64 + 32);
#pragma unroll
  for (int nt = 0; nt < 4; ++nt) {
    v8f c = {};
#pragma unroll
    for (int ki = 0; ki < 2; ++ki) c = wmma32(a2[ki], loadBfrag(W2F, nt * 2 + ki), c);
    const int n = nt * 16 + (lane & 15);
    const float bias = b2[n];
    storeC(c, [&](int r, float v) {
      nodeF16[(size_t)(r0 + r) * 64 + n] = (_Float16)(v + bias);
    });
  }
}

// ---------------- final edge head: [src|dst|edge_attr] -> 128 -> 64 -> 1 ----------------
__global__ __launch_bounds__(32) void final_edge_kernel(
    const _Float16* __restrict__ nodeF16, const float* __restrict__ edge_attr,
    const int* __restrict__ ei,
    const _Float16* __restrict__ Wem1F, const float* __restrict__ em_b1,
    const _Float16* __restrict__ Wem2F, const float* __restrict__ em_b2,
    const float* __restrict__ em_w3, const float* __restrict__ em_b3,
    float* __restrict__ out)
{
  __shared__ __align__(32) _Float16 sH1[16 * 128];
  __shared__ __align__(32) _Float16 sH2[16 * 64];
  const int lane = threadIdx.x;
  const int e0 = blockIdx.x * 16;
  const int eRow = e0 + (lane & 15);
  __builtin_prefetch(Wem1F, 0, 3);
  __builtin_prefetch(Wem2F, 0, 3);
  const int s = ei[eRow];
  const int d = ei[EE + eRow];
  const _Float16* ns = nodeF16 + (size_t)s * 64;
  const _Float16* nd = nodeF16 + (size_t)d * 64;
  const float* ea = edge_attr + (size_t)eRow * 6;
  v16h af[5];  // K = 134 padded to 160
  af[0] = loadA_row32(ns);  af[1] = loadA_row32(ns + 32);
  af[2] = loadA_row32(nd);  af[3] = loadA_row32(nd + 32);
  af[4] = loadA([&](int ml, int k) -> _Float16 {
    return (k < 6) ? (_Float16)ea[k] : (_Float16)0.f;  // kk = 128+k
  });
#pragma unroll
  for (int nt = 0; nt < 8; ++nt) {
    v8f c = {};
#pragma unroll
    for (int ki = 0; ki < 5; ++ki) c = wmma32(af[ki], loadBfrag(Wem1F, nt * 5 + ki), c);
    const int n = nt * 16 + (lane & 15);
    const float bias = em_b1[n];
    storeC(c, [&](int r, float v) {
      float h = v + bias;
      sH1[r * 128 + n] = (_Float16)(h > 0.f ? h : 0.f);
    });
  }
  __syncthreads();
  v16h a2[4];
#pragma unroll
  for (int ki = 0; ki < 4; ++ki)
    a2[ki] = loadA_row32(sH1 + (lane & 15) * 128 + ki * 32);
#pragma unroll
  for (int nt = 0; nt < 4; ++nt) {
    v8f c = {};
#pragma unroll
    for (int ki = 0; ki < 4; ++ki) c = wmma32(a2[ki], loadBfrag(Wem2F, nt * 4 + ki), c);
    const int n = nt * 16 + (lane & 15);
    const float bias = em_b2[n];
    storeC(c, [&](int r, float v) {
      float h = v + bias;
      sH2[r * 64 + n] = (_Float16)(h > 0.f ? h : 0.f);
    });
  }
  __syncthreads();
  if (lane < 16) {
    float acc = em_b3[0];
#pragma unroll
    for (int k = 0; k < 64; ++k) acc += (float)sH2[lane * 64 + k] * em_w3[k];
    out[e0 + lane] = acc;
  }
}

// ---------------- host launcher ----------------
extern "C" void kernel_launch(void* const* d_in, const int* in_sizes, int n_in,
                              void* d_out, int out_size, void* d_ws, size_t ws_size,
                              hipStream_t stream)
{
  (void)in_sizes; (void)n_in; (void)out_size; (void)ws_size;
  const float* x         = (const float*)d_in[0];
  const int*   ei        = (const int*)d_in[1];
  const float* edge_attr = (const float*)d_in[2];
  const float* crop      = (const float*)d_in[3];
  const float* cw0       = (const float*)d_in[4];
  const float* cb0       = (const float*)d_in[5];
  const float* cw        = (const float*)d_in[6];
  const float* cb        = (const float*)d_in[7];
  const float* nu_w      = (const float*)d_in[8];
  const float* nu_b      = (const float*)d_in[9];
  const float* nu_g      = (const float*)d_in[10];
  const float* nu_be     = (const float*)d_in[11];
  const float* eu_w      = (const float*)d_in[12];
  const float* eu_b      = (const float*)d_in[13];
  const float* eu_g      = (const float*)d_in[14];
  const float* eu_be     = (const float*)d_in[15];
  const float* ie_w1     = (const float*)d_in[16];
  const float* ie_b1     = (const float*)d_in[17];
  const float* ie_w2     = (const float*)d_in[18];
  const float* ie_b2     = (const float*)d_in[19];
  const float* in_w1     = (const float*)d_in[20];
  const float* in_b1     = (const float*)d_in[21];
  const float* in_w2     = (const float*)d_in[22];
  const float* in_b2     = (const float*)d_in[23];
  const float* em_w1     = (const float*)d_in[24];
  const float* em_b1     = (const float*)d_in[25];
  const float* em_w2     = (const float*)d_in[26];
  const float* em_b2     = (const float*)d_in[27];
  const float* em_w3     = (const float*)d_in[28];
  const float* em_b3     = (const float*)d_in[29];

  char* ws = (char*)d_ws;
  size_t off = 0;
  auto carve = [&](size_t bytes) -> void* {
    void* p = ws + off;
    off = (off + bytes + 255) & ~(size_t)255;
    return p;
  };
  _Float16* nodeF16 = (_Float16*)carve((size_t)NN * 64 * 2);
  _Float16* edgeF16 = (_Float16*)carve((size_t)EE * 64 * 2);
  float*    agg     = (float*)carve((size_t)NN * 64 * 4);
  _Float16* Wc0F    = (_Float16*)carve(2048 * 2);
  _Float16* WcF     = (_Float16*)carve(81920 * 2);
  _Float16* Wie1F   = (_Float16*)carve(49152 * 2);
  _Float16* Wie2F   = (_Float16*)carve(16384 * 2);
  _Float16* Win1F   = (_Float16*)carve(32768 * 2);
  _Float16* Win2F   = (_Float16*)carve(16384 * 2);
  _Float16* Wem1F   = (_Float16*)carve(20480 * 2);
  _Float16* Wem2F   = (_Float16*)carve(8192 * 2);

  prep_weights<<<dim3(888), dim3(256), 0, stream>>>(
      cw0, cw, ie_w1, ie_w2, in_w1, in_w2, em_w1, em_w2,
      Wc0F, WcF, Wie1F, Wie2F, Win1F, Win2F, Wem1F, Wem2F);

  conv_node_kernel<<<dim3(NN), dim3(256), 0, stream>>>(
      crop, x, cb0, cb, nu_w, nu_b, nu_g, nu_be, Wc0F, WcF, nodeF16);

  edge_update_kernel<<<dim3((EE + 255) / 256), dim3(256), 0, stream>>>(
      edge_attr, eu_w, eu_b, eu_g, eu_be, edgeF16);

  for (int l = 0; l < 4; ++l) {
    hipMemsetAsync(agg, 0, (size_t)NN * 64 * 4, stream);
    msg_kernel<<<dim3(EE / 16), dim3(32), 0, stream>>>(
        nodeF16, edgeF16, ei,
        Wie1F + (size_t)l * 12288, ie_b1 + l * 64,
        Wie2F + (size_t)l * 4096, ie_b2 + l * 64, agg);
    node_mlp_kernel<<<dim3(NN / 16), dim3(32), 0, stream>>>(
        nodeF16, agg,
        Win1F + (size_t)l * 8192, in_b1 + l * 64,
        Win2F + (size_t)l * 4096, in_b2 + l * 64);
  }

  final_edge_kernel<<<dim3(EE / 16), dim3(32), 0, stream>>>(
      nodeF16, edge_attr, ei, Wem1F, em_b1, Wem2F, em_b2, em_w3, em_b3,
      (float*)d_out);
}
